// GraphAttentionAutoencoder_51951924412525
// MI455X (gfx1250) — compile-verified, matrix operand order checked by reading, not run on hardware
//
#include <hip/hip_runtime.h>

typedef __bf16 v8bf  __attribute__((ext_vector_type(8)));
typedef __bf16 v16bf __attribute__((ext_vector_type(16)));
typedef float  v8f   __attribute__((ext_vector_type(8)));

constexpr int kN    = 50000;
constexpr int kE    = 800000;
constexpr int kIN   = 3000;
constexpr int kHID  = 512;
constexpr int kLAT  = 30;
constexpr int kKPAD = 3008;           // IN padded to multiple of 32 (= 188*16)
constexpr int kET   = kE + kN;        // edges + self loops
constexpr float kSLOPE = 0.2f;

// ---------- helpers ----------
__device__ __forceinline__ int fenc(float f) {            // monotonic float->int
  int i = __float_as_int(f);
  return i >= 0 ? i : (i ^ 0x7FFFFFFF);
}
__device__ __forceinline__ float fdec(int i) {            // involution
  i = i >= 0 ? i : (i ^ 0x7FFFFFFF);
  return __int_as_float(i);
}
__device__ __forceinline__ v16bf cat8(v8bf lo, v8bf hi) {
  return __builtin_shufflevector(lo, hi, 0,1,2,3,4,5,6,7,8,9,10,11,12,13,14,15);
}
__device__ __forceinline__ float elu1(float v) {
  return v > 0.f ? v : (__expf(v) - 1.f);
}

// ---------- 0: weight conversion: Wb (HID x KPAD bf16), Wt (KPAD x HID bf16, transposed) ----------
__global__ void convert_w_k(const float* __restrict__ W,
                            unsigned short* __restrict__ Wb_us,
                            unsigned short* __restrict__ Wt_us) {
  __bf16* Wb = (__bf16*)Wb_us;
  __bf16* Wt = (__bf16*)Wt_us;
  int i = blockIdx.x * 256 + threadIdx.x;
  if (i >= kHID * kKPAD) return;
  int hh = i / kKPAD, k = i % kKPAD;
  float v = (k < kIN) ? W[(size_t)hh * kIN + k] : 0.f;
  __bf16 bv = (__bf16)v;
  Wb[(size_t)hh * kKPAD + k] = bv;
  Wt[(size_t)k * kHID + hh]  = bv;     // rows k>=3000 are zero-padded columns
}

// ---------- 1: encoder GEMM  h = elu(x @ W^T + b1) ----------
__global__ __launch_bounds__(256) void enc_gemm_k(const float* __restrict__ x,
                                                  const unsigned short* __restrict__ Wb_us,
                                                  const float* __restrict__ b1,
                                                  float* __restrict__ h) {
  const __bf16* Wb = (const __bf16*)Wb_us;
  __shared__ __bf16 sx[16 * 520];                 // 16 rows, 512-chunk + pad
  const int tid  = threadIdx.x;
  const int lane = tid & 31;
  const int wv   = tid >> 5;                      // wave id 0..7
  const int node0 = blockIdx.x * 16;

  v8f zero = {};
  v8f acc[4] = {zero, zero, zero, zero};

  const int row  = lane & 15;
  const int koff = (lane < 16) ? 0 : 8;

  for (int kc = 0; kc < kKPAD; kc += 512) {
    const int klen = (kKPAD - kc) < 512 ? (kKPAD - kc) : 512;   // 512 or 448
    __syncthreads();
    for (int i = tid; i < 16 * klen; i += 256) {
      int r = i / klen, k = i % klen;
      int kg = kc + k;
      float v = (kg < kIN) ? x[(size_t)(node0 + r) * kIN + kg] : 0.f;
      sx[r * 520 + k] = (__bf16)v;
    }
    __syncthreads();

    for (int kk = 0; kk < klen; kk += 32) {
      v8bf alo = *(const v8bf*)&sx[row * 520 + kk + koff];
      v8bf ahi = *(const v8bf*)&sx[row * 520 + kk + koff + 16];
      v16bf a = cat8(alo, ahi);
      const int kg = kc + kk;
#pragma unroll
      for (int j = 0; j < 4; j++) {
        int hid = (wv * 4 + j) * 16 + (lane & 15);
        const __bf16* bp = Wb + (size_t)hid * kKPAD + kg + koff;
        v8bf blo = *(const v8bf*)bp;
        v8bf bhi = *(const v8bf*)(bp + 16);
        v16bf b = cat8(blo, bhi);
        acc[j] = __builtin_amdgcn_wmma_f32_16x16x32_bf16(
            false, a, false, b, (short)0, acc[j], false, false);
      }
    }
  }

  const int mbase = (lane < 16) ? 0 : 8;
#pragma unroll
  for (int j = 0; j < 4; j++) {
    int hid = (wv * 4 + j) * 16 + (lane & 15);
    float bias = b1[hid];
#pragma unroll
    for (int v = 0; v < 8; v++) {
      float val = acc[j][v] + bias;
      h[(size_t)(node0 + mbase + v) * kHID + hid] = elu1(val);
    }
  }
}

// ---------- 2: GAT projection  xh = h @ gat_w  (512 -> 30) ----------
__global__ void gat_proj_k(const float* __restrict__ h, const float* __restrict__ gw,
                           float* __restrict__ xh) {
  int i = blockIdx.x * 256 + threadIdx.x;
  if (i >= kN * kLAT) return;
  int n = i / kLAT, c = i % kLAT;
  const float* hr = h + (size_t)n * kHID;
  float s = 0.f;
#pragma unroll 4
  for (int k = 0; k < kHID; k++) s += hr[k] * gw[k * kLAT + c];
  xh[i] = s;
}

// ---------- 3: per-node attention scores ----------
__global__ void scores_k(const float* __restrict__ xh, const float* __restrict__ att,
                         float* __restrict__ si, float* __restrict__ sj) {
  int n = blockIdx.x * 256 + threadIdx.x;
  if (n >= kN) return;
  const float* xr = xh + (size_t)n * kLAT;
  float a = 0.f, b = 0.f;
#pragma unroll
  for (int c = 0; c < kLAT; c++) { a += xr[c] * att[c]; b += xr[c] * att[kLAT + c]; }
  si[n] = a; sj[n] = b;
}

// ---------- 4: init segment buffers ----------
__global__ void seg_init_k(int* __restrict__ amax, float* __restrict__ denom,
                           float* __restrict__ numer) {
  int i = blockIdx.x * 256 + threadIdx.x;
  int n = i >> 5, c = i & 31;
  if (n >= kN) return;
  if (c < kLAT)       numer[(size_t)n * kLAT + c] = 0.f;
  else if (c == 30)   denom[n] = 0.f;
  else                amax[n]  = (int)0x80000000;   // < enc(any float)
}

// ---------- 5: edge pass 1 — leaky_relu, *eattr, segment max ----------
__global__ void edge_max_k(const int* __restrict__ ei, const float* __restrict__ ea,
                           const float* __restrict__ si, const float* __restrict__ sj,
                           float* __restrict__ alpha, int* __restrict__ amax) {
  int e = blockIdx.x * 256 + threadIdx.x;
  if (e >= kET) return;
  int s, d; float w;
  if (e < kE) { s = ei[e]; d = ei[kE + e]; w = ea[e]; }
  else        { s = d = e - kE; w = 1.f; }
  float a = si[d] + sj[s];
  a = a > 0.f ? a : kSLOPE * a;
  a *= w;
  alpha[e] = a;
  atomicMax(&amax[d], fenc(a));
}

// ---------- 6: edge pass 2 — exp, segment sums (numerator + denominator) ----------
__global__ void edge_sum_k(const int* __restrict__ ei, const float* __restrict__ alpha,
                           const int* __restrict__ amax, const float* __restrict__ xh,
                           float* __restrict__ numer, float* __restrict__ denom) {
  long long gid = (long long)blockIdx.x * 256 + threadIdx.x;
  int e = (int)(gid >> 5);
  int c = (int)(gid & 31);
  if (e >= kET) return;
  int s, d;
  if (e < kE) { s = ei[e]; d = ei[kE + e]; }
  else        { s = d = e - kE; }
  float wv = __expf(alpha[e] - fdec(amax[d]));
  if (c < kLAT)      atomicAdd(&numer[(size_t)d * kLAT + c], wv * xh[(size_t)s * kLAT + c]);
  else if (c == 31)  atomicAdd(&denom[d], wv);
}

// ---------- 7: finalize z = elu(numer/denom + gat_b) ----------
__global__ void finalize_z_k(const float* __restrict__ numer, const float* __restrict__ denom,
                             const float* __restrict__ gb, float* __restrict__ z) {
  int i = blockIdx.x * 256 + threadIdx.x;
  if (i >= kN * kLAT) return;
  int n = i / kLAT, c = i % kLAT;
  float v = numer[i] / (denom[n] + 1e-16f) + gb[c];
  z[i] = elu1(v);
}

// ---------- 8: decoder layer 1  hd = elu(z @ dec_w1^T + b), stored bf16 ----------
__global__ void dec1_k(const float* __restrict__ z, const float* __restrict__ dw,
                       const float* __restrict__ db, unsigned short* __restrict__ hdb_us) {
  __bf16* hdb = (__bf16*)hdb_us;
  int i = blockIdx.x * 256 + threadIdx.x;
  if (i >= kN * kHID) return;
  int n = i >> 9, hh = i & 511;
  const float* zr = z + (size_t)n * kLAT;
  const float* wr = dw + (size_t)hh * kLAT;
  float s = db[hh];
#pragma unroll
  for (int c = 0; c < kLAT; c++) s += zr[c] * wr[c];
  hdb[i] = (__bf16)elu1(s);
}

// ---------- 9: decoder GEMM  x_recon = hd @ enc_w1 + dec_b2 ----------
__global__ __launch_bounds__(256) void dec_gemm_k(const unsigned short* __restrict__ hdb_us,
                                                  const unsigned short* __restrict__ Wt_us,
                                                  const float* __restrict__ b2,
                                                  float* __restrict__ xr) {
  const __bf16* hdb = (const __bf16*)hdb_us;
  const __bf16* Wt  = (const __bf16*)Wt_us;
  __shared__ __bf16 sh[16 * 520];
  const int tid  = threadIdx.x;
  const int lane = tid & 31;
  const int wv   = tid >> 5;
  const int node0 = blockIdx.x * 16;

  for (int i = tid; i < 16 * kHID; i += 256) {
    int r = i >> 9, k = i & 511;
    sh[r * 520 + k] = hdb[(size_t)(node0 + r) * kHID + k];
  }
  __syncthreads();

  const int row  = lane & 15;
  const int koff = (lane < 16) ? 0 : 8;
  const int mbase = (lane < 16) ? 0 : 8;
  v8f zero = {};

  for (int t = wv; t < kKPAD / 16; t += 8) {         // 188 column tiles
    v8f acc = zero;
    const int col = t * 16 + (lane & 15);
#pragma unroll 4
    for (int kk = 0; kk < kHID; kk += 32) {
      v8bf alo = *(const v8bf*)&sh[row * 520 + kk + koff];
      v8bf ahi = *(const v8bf*)&sh[row * 520 + kk + koff + 16];
      v16bf a = cat8(alo, ahi);
      const __bf16* bp = Wt + (size_t)col * kHID + kk + koff;
      v8bf blo = *(const v8bf*)bp;
      v8bf bhi = *(const v8bf*)(bp + 16);
      v16bf b = cat8(blo, bhi);
      acc = __builtin_amdgcn_wmma_f32_16x16x32_bf16(
          false, a, false, b, (short)0, acc, false, false);
    }
    if (col < kIN) {
      float bias = b2[col];
#pragma unroll
      for (int v = 0; v < 8; v++)
        xr[(size_t)(node0 + mbase + v) * kIN + col] = acc[v] + bias;
    }
  }
}

// ---------- launch ----------
extern "C" void kernel_launch(void* const* d_in, const int* in_sizes, int n_in,
                              void* d_out, int out_size, void* d_ws, size_t ws_size,
                              hipStream_t stream) {
  (void)in_sizes; (void)n_in; (void)out_size; (void)ws_size;
  const float* x      = (const float*)d_in[0];
  const int*   ei     = (const int*)  d_in[1];
  const float* ea     = (const float*)d_in[2];
  const float* enc_w1 = (const float*)d_in[3];
  const float* enc_b1 = (const float*)d_in[4];
  const float* gat_w  = (const float*)d_in[5];
  const float* gat_att= (const float*)d_in[6];
  const float* gat_b  = (const float*)d_in[7];
  const float* dec_w1 = (const float*)d_in[8];
  const float* dec_b1 = (const float*)d_in[9];
  const float* dec_b2 = (const float*)d_in[10];

  float* xr = (float*)d_out;                      // (N, IN)
  float* z  = xr + (size_t)kN * kIN;              // (N, LAT)

  // workspace carve (256B aligned)
  char* p = (char*)d_ws;
  auto carve = [&](size_t bytes) -> void* {
    void* r = (void*)p;
    p += (bytes + 255) & ~(size_t)255;
    return r;
  };
  unsigned short* Wb   = (unsigned short*)carve((size_t)kHID * kKPAD * 2);
  unsigned short* Wt   = (unsigned short*)carve((size_t)kKPAD * kHID * 2);
  float*          h    = (float*)         carve((size_t)kN * kHID * 4);
  float*          xh   = (float*)         carve((size_t)kN * kLAT * 4);
  float*          si   = (float*)         carve((size_t)kN * 4);
  float*          sj   = (float*)         carve((size_t)kN * 4);
  int*            amax = (int*)           carve((size_t)kN * 4);
  float*          den  = (float*)         carve((size_t)kN * 4);
  float*          num  = (float*)         carve((size_t)kN * kLAT * 4);
  float*          alp  = (float*)         carve((size_t)kET * 4);
  unsigned short* hdb  = (unsigned short*)h;      // alias: h dead after gat_proj

  convert_w_k<<<(kHID * kKPAD + 255) / 256, 256, 0, stream>>>(enc_w1, Wb, Wt);
  enc_gemm_k<<<kN / 16, 256, 0, stream>>>(x, Wb, enc_b1, h);
  gat_proj_k<<<(kN * kLAT + 255) / 256, 256, 0, stream>>>(h, gat_w, xh);
  scores_k<<<(kN + 255) / 256, 256, 0, stream>>>(xh, gat_att, si, sj);
  seg_init_k<<<(kN * 32 + 255) / 256, 256, 0, stream>>>(amax, den, num);
  edge_max_k<<<(kET + 255) / 256, 256, 0, stream>>>(ei, ea, si, sj, alp, amax);
  edge_sum_k<<<(int)(((long long)kET * 32 + 255) / 256), 256, 0, stream>>>(ei, alp, amax, xh, num, den);
  finalize_z_k<<<(kN * kLAT + 255) / 256, 256, 0, stream>>>(num, den, gat_b, z);
  dec1_k<<<(kN * kHID + 255) / 256, 256, 0, stream>>>(z, dec_w1, dec_b1, hdb);
  dec_gemm_k<<<kN / 16, 256, 0, stream>>>(hdb, Wt, dec_b2, xr);
}
